// SocialEncoder_59090160058724
// MI455X (gfx1250) — matrix operand order repaired and checked by reading, not compiled, not run on hardware
//
#include <hip/hip_runtime.h>

#define BB 256
#define SS 32
#define LL 20
#define DD 512
#define NROWS (BB + BB * SS)   // 8448 agg rows (self first, then social b*S+s)
#define IN2D  (2 * DD)         // 1024

typedef __attribute__((ext_vector_type(16))) __bf16        v16bf;
typedef __attribute__((ext_vector_type(8)))  float         v8f;
typedef __attribute__((ext_vector_type(8)))  unsigned int  v8u;

__device__ __forceinline__ unsigned short f32_to_bf16(float f) {
    unsigned int u = __builtin_bit_cast(unsigned int, f);
    unsigned int r = u + 0x7FFFu + ((u >> 16) & 1u);   // round-to-nearest-even
    return (unsigned short)(r >> 16);
}

__device__ __forceinline__ v16bf load_frag_a(const unsigned short* row, int k0, int sel) {
    // A fragment: lanes 0-15 -> K {0..7,16..23}; lanes 16-31 -> K {8..15,24..31}
    uint4 a0 = *(const uint4*)(row + k0 + sel * 8);
    uint4 a1 = *(const uint4*)(row + k0 + 16 + sel * 8);
    v8u au = { a0.x, a0.y, a0.z, a0.w, a1.x, a1.y, a1.z, a1.w };
    return __builtin_bit_cast(v16bf, au);
}

__device__ __forceinline__ v16bf load_frag_b(const unsigned short* row, int k0, int sel) {
    // B fragment: lane column lr; lanes 0-15 -> K {0..7,16..23}; lanes 16-31 -> K {8..15,24..31}
    uint4 b0 = *(const uint4*)(row + k0 + sel * 8);
    uint4 b1 = *(const uint4*)(row + k0 + 16 + sel * 8);
    v8u bu = { b0.x, b0.y, b0.z, b0.w, b1.x, b1.y, b1.z, b1.w };
    return __builtin_bit_cast(v16bf, bu);
}

// ---------------------------------------------------------------- cvt f32->bf16
__global__ void cvt_kernel(const float* __restrict__ src,
                           unsigned short* __restrict__ dst, int n) {
    int i = blockIdx.x * blockDim.x + threadIdx.x;
    if (i < n) dst[i] = f32_to_bf16(src[i]);
}

// ------------------------------------------------- masked mean over L -> bf16 rows
// row r < 256: self (b = r); else social idx = r-256 = b*S+s
__global__ void aggregate_kernel(const float* __restrict__ self_u,
                                 const float* __restrict__ self_i,
                                 const float* __restrict__ self_m,
                                 const float* __restrict__ soc_u,
                                 const float* __restrict__ soc_i,
                                 const float* __restrict__ soc_m,
                                 unsigned short* __restrict__ agg /*[NROWS][1024]*/) {
    int r = blockIdx.x;
    const float *U, *I, *Mk;
    if (r < BB) {
        U = self_u + (size_t)r * LL * DD;
        I = self_i + (size_t)r * LL * DD;
        Mk = self_m + (size_t)r * LL;
    } else {
        int idx = r - BB;
        U = soc_u + (size_t)idx * LL * DD;
        I = soc_i + (size_t)idx * LL * DD;
        Mk = soc_m + (size_t)idx * LL;
    }
    __shared__ float mk[LL];
    int tid = threadIdx.x;                 // 256 threads
    if (tid < LL) mk[tid] = Mk[tid];
    __syncthreads();
    float msum = 0.f;
#pragma unroll
    for (int l = 0; l < LL; ++l) msum += mk[l];
    float inv = 1.0f / (msum + 1e-4f);

    int c4 = tid * 4;                      // column group within concat dim 1024
    const float* src = (c4 < DD) ? (U + c4) : (I + (c4 - DD));
    float ax = 0.f, ay = 0.f, az = 0.f, aw = 0.f;
#pragma unroll
    for (int l = 0; l < LL; ++l) {
        float4 v = *(const float4*)(src + (size_t)l * DD);
        float m = mk[l];
        ax += m * v.x; ay += m * v.y; az += m * v.z; aw += m * v.w;
    }
    ax *= inv; ay *= inv; az *= inv; aw *= inv;
    uint2 o;
    o.x = (unsigned int)f32_to_bf16(ax) | ((unsigned int)f32_to_bf16(ay) << 16);
    o.y = (unsigned int)f32_to_bf16(az) | ((unsigned int)f32_to_bf16(aw) << 16);
    *(uint2*)(agg + (size_t)r * IN2D + c4) = o;
}

// --------------------------------------------- WMMA bf16 GEMM + bias + ReLU
// C[M][N] = relu(A[M][K] * B^T + bias), Bm stored as [N][K] (B column-major).
// Register-blocked: each wave computes a 64x32 strip (4x2 tiles of 16x16),
// 8 v_wmma_f32_16x16x32_bf16 per K-step from 6 fragment loads.
#define MT 4
#define NT 2
__global__ void wmma_gemm_bias_relu(const unsigned short* __restrict__ A,
                                    const unsigned short* __restrict__ Bm,
                                    const float* __restrict__ bias,
                                    float* __restrict__ Cf,
                                    unsigned short* __restrict__ Cb,
                                    int M, int N, int K) {
    int mblocks = M / (16 * MT);
    int nblocks = N / (16 * NT);
    int wid = blockIdx.x * (blockDim.x >> 5) + (threadIdx.x >> 5);
    if (wid >= mblocks * nblocks) return;
    int mb = wid / nblocks;
    int nb = wid - mb * nblocks;
    int lane = threadIdx.x & 31;
    int lr   = lane & 15;
    int sel  = lane >> 4;

    const unsigned short* Arow[MT];
#pragma unroll
    for (int mi = 0; mi < MT; ++mi)
        Arow[mi] = A + (size_t)(mb * (16 * MT) + mi * 16 + lr) * K;
    const unsigned short* Brow[NT];
#pragma unroll
    for (int ni = 0; ni < NT; ++ni)
        Brow[ni] = Bm + (size_t)(nb * (16 * NT) + ni * 16 + lr) * K;

    v8f acc[MT][NT];
#pragma unroll
    for (int mi = 0; mi < MT; ++mi)
#pragma unroll
        for (int ni = 0; ni < NT; ++ni)
            acc[mi][ni] = (v8f){};

    for (int k0 = 0; k0 < K; k0 += 32) {
        v16bf av[MT], bv[NT];
#pragma unroll
        for (int mi = 0; mi < MT; ++mi) av[mi] = load_frag_a(Arow[mi], k0, sel);
#pragma unroll
        for (int ni = 0; ni < NT; ++ni) bv[ni] = load_frag_b(Brow[ni], k0, sel);
#pragma unroll
        for (int mi = 0; mi < MT; ++mi)
#pragma unroll
            for (int ni = 0; ni < NT; ++ni)
                acc[mi][ni] = __builtin_amdgcn_wmma_f32_16x16x32_bf16(
                    /*neg_a=*/false, av[mi], /*neg_b=*/false, bv[ni],
                    /*c_mod=*/(short)0, acc[mi][ni],
                    /*reuse_a=*/false, /*reuse_b=*/false);
    }

#pragma unroll
    for (int ni = 0; ni < NT; ++ni) {
        int n = nb * (16 * NT) + ni * 16 + lr;
        float bvv = bias ? bias[n] : 0.0f;
#pragma unroll
        for (int mi = 0; mi < MT; ++mi) {
#pragma unroll
            for (int v = 0; v < 8; ++v) {
                int m = mb * (16 * MT) + mi * 16 + sel * 8 + v; // C/D: VGPR v -> row sel*8+v
                float r = acc[mi][ni][v] + bvv;
                r = r > 0.f ? r : 0.f;
                if (Cf) Cf[(size_t)m * N + n] = r;
                if (Cb) Cb[(size_t)m * N + n] = f32_to_bf16(r);
            }
        }
    }
}

// --------------------- per-user softmax attention over 32 neighbors, f32
__global__ void attention_kernel(const float* __restrict__ embd /*[NROWS][512]*/,
                                 float* __restrict__ attn_f /*[256][512]*/,
                                 unsigned short* __restrict__ attn_bf) {
    int b = blockIdx.x;
    const float* selfv = embd + (size_t)b * DD;                 // self rows 0..255
    const float* soc   = embd + (size_t)(BB + b * SS) * DD;     // 32 neighbor rows
    __shared__ float sc[SS];
    __shared__ float att[SS];
    int tid = threadIdx.x;              // 256 threads = 8 waves
    int wave = tid >> 5, lane = tid & 31;

#pragma unroll
    for (int j = 0; j < 4; ++j) {       // each wave: 4 neighbors
        int s = wave * 4 + j;
        const float* sv = soc + (size_t)s * DD;
        float p = 0.f;
        for (int d = lane; d < DD; d += 32) p += selfv[d] * sv[d];
#pragma unroll
        for (int off = 16; off >= 1; off >>= 1) p += __shfl_xor(p, off, 32);
        if (lane == 0) sc[s] = p;
    }
    __syncthreads();
    if (tid < SS) {                     // softmax over S inside one wave
        float v = sc[tid];
        float mx = v;
#pragma unroll
        for (int off = 16; off >= 1; off >>= 1) mx = fmaxf(mx, __shfl_xor(mx, off, 32));
        float e = __expf(v - mx);
        float ssum = e;
#pragma unroll
        for (int off = 16; off >= 1; off >>= 1) ssum += __shfl_xor(ssum, off, 32);
        att[tid] = e / ssum;
    }
    __syncthreads();
    for (int d = tid; d < DD; d += 256) {
        float a = 0.f;
#pragma unroll 8
        for (int s = 0; s < SS; ++s) a += att[s] * soc[(size_t)s * DD + d];
        attn_f[(size_t)b * DD + d] = a;
        attn_bf[(size_t)b * DD + d] = f32_to_bf16(a);
    }
}

// ---- scores = w2 . t[b], softmax over batch, out[d] = sum_b att[b]*embd[b][d]
__global__ void rate_finalize(const float* __restrict__ t /*[256][512]*/,
                              const float* __restrict__ embd /*[256][512]*/,
                              const float* __restrict__ w2 /*[512]*/,
                              float* __restrict__ out /*[512]*/) {
    __shared__ float buf[BB];
    __shared__ float att[BB];
    int tid = threadIdx.x;              // 256 threads, thread == batch element
    const float* tb = t + (size_t)tid * DD;
    float s = 0.f;
    for (int d = 0; d < DD; ++d) s += w2[d] * tb[d];
    buf[tid] = s; __syncthreads();
    for (int st = 128; st >= 1; st >>= 1) {
        if (tid < st) buf[tid] = fmaxf(buf[tid], buf[tid + st]);
        __syncthreads();
    }
    float mx = buf[0]; __syncthreads();
    float e = __expf(s - mx);
    buf[tid] = e; __syncthreads();
    for (int st = 128; st >= 1; st >>= 1) {
        if (tid < st) buf[tid] += buf[tid + st];
        __syncthreads();
    }
    float inv = 1.0f / buf[0];
    att[tid] = e * inv; __syncthreads();
    for (int d = tid; d < DD; d += 256) {
        float a = 0.f;
        for (int b = 0; b < BB; ++b) a += att[b] * embd[(size_t)b * DD + d];
        out[d] = a;
    }
}

extern "C" void kernel_launch(void* const* d_in, const int* in_sizes, int n_in,
                              void* d_out, int out_size, void* d_ws, size_t ws_size,
                              hipStream_t stream) {
    (void)in_sizes; (void)n_in; (void)out_size; (void)ws_size;
    const float* self_u = (const float*)d_in[0];
    const float* self_i = (const float*)d_in[1];
    const float* self_m = (const float*)d_in[2];
    const float* soc_u  = (const float*)d_in[3];
    const float* soc_i  = (const float*)d_in[4];
    const float* soc_m  = (const float*)d_in[5];
    const float* w_user = (const float*)d_in[6];   // [512][1024]
    const float* b_user = (const float*)d_in[7];   // [512]
    const float* w1     = (const float*)d_in[8];   // [512][512]
    const float* b1     = (const float*)d_in[9];   // [512]
    const float* w2     = (const float*)d_in[10];  // [512]
    float* out = (float*)d_out;                    // [1024] = se ++ so

    char* ws = (char*)d_ws;
    size_t off = 0;
    auto take = [&](size_t bytes) { size_t p = off; off += (bytes + 255) & ~(size_t)255; return p; };
    unsigned short* wuser_bf = (unsigned short*)(ws + take((size_t)DD * IN2D * 2));
    unsigned short* w1_bf    = (unsigned short*)(ws + take((size_t)DD * DD * 2));
    unsigned short* agg_bf   = (unsigned short*)(ws + take((size_t)NROWS * IN2D * 2));
    float*          embd_f   = (float*)(ws + take((size_t)NROWS * DD * 4));
    unsigned short* embd_bf  = (unsigned short*)(ws + take((size_t)NROWS * DD * 2));
    float*          attn_f   = (float*)(ws + take((size_t)BB * DD * 4));
    unsigned short* attn_bf  = (unsigned short*)(ws + take((size_t)BB * DD * 2));
    float*          t_f      = (float*)(ws + take((size_t)BB * DD * 4));

    // 1) weights -> bf16
    cvt_kernel<<<(DD * IN2D + 255) / 256, 256, 0, stream>>>(w_user, wuser_bf, DD * IN2D);
    cvt_kernel<<<(DD * DD + 255) / 256, 256, 0, stream>>>(w1, w1_bf, DD * DD);

    // 2) masked mean over history (bandwidth-bound stage, ~692 MB streamed)
    aggregate_kernel<<<NROWS, 256, 0, stream>>>(self_u, self_i, self_m,
                                                soc_u, soc_i, soc_m, agg_bf);

    // 3) GEMM1: [8448x1024]@[1024x512] + bias + relu (WMMA bf16, 64x32/wave)
    {
        int waves = (NROWS / (16 * MT)) * (DD / (16 * NT));   // 132*16 = 2112
        wmma_gemm_bias_relu<<<(waves + 3) / 4, 128, 0, stream>>>(
            agg_bf, wuser_bf, b_user, embd_f, embd_bf, NROWS, DD, IN2D);
    }

    // 4) attention over neighbors
    attention_kernel<<<BB, 256, 0, stream>>>(embd_f, attn_f, attn_bf);

    // 5) se = aggregate_rate(self_embd)
    {
        int waves = (BB / (16 * MT)) * (DD / (16 * NT));      // 4*16 = 64
        wmma_gemm_bias_relu<<<(waves + 3) / 4, 128, 0, stream>>>(
            embd_bf, w1_bf, b1, t_f, nullptr, BB, DD, DD);
        rate_finalize<<<1, BB, 0, stream>>>(t_f, embd_f, w2, out);
        // 6) so = aggregate_rate(social_attn)
        wmma_gemm_bias_relu<<<(waves + 3) / 4, 128, 0, stream>>>(
            attn_bf, w1_bf, b1, t_f, nullptr, BB, DD, DD);
        rate_finalize<<<1, BB, 0, stream>>>(t_f, attn_f, w2, out + DD);
    }
}
